// Head_54004918780372
// MI455X (gfx1250) — matrix operand order repaired and checked by reading, not compiled, not run on hardware
//
#include <hip/hip_runtime.h>

typedef __attribute__((ext_vector_type(16))) __bf16 v16bf;
typedef __attribute__((ext_vector_type(8)))  __bf16 v8bf;
typedef __attribute__((ext_vector_type(4)))  __bf16 v4bf;
typedef __attribute__((ext_vector_type(8)))  float  v8f;

#define B_ 4
#define T_ 2048
#define C_ 1024
#define H_ 128
#define SOFTMAX_SCALE 0.08838834764831845f  // 1/sqrt(128)

__device__ __forceinline__ v16bf cat16(v8bf a, v8bf b) {
    return __builtin_shufflevector(a, b, 0,1,2,3,4,5,6,7,8,9,10,11,12,13,14,15);
}

__device__ __forceinline__ v8f wmma_bf16(v16bf a, v16bf b, v8f c) {
    // D(16x16,f32) = A(16x32,bf16) * B(32x16,bf16) + C
    return __builtin_amdgcn_wmma_f32_16x16x32_bf16(
        /*neg_a=*/false, a, /*neg_b=*/false, b,
        /*c_mod=*/(short)0, c, /*reuse_a=*/false, /*reuse_b=*/false);
}

// ---------------------------------------------------------------------------
// Kernel 0a: x (f32) -> xb (bf16), flat, vectorized x4
// ---------------------------------------------------------------------------
__global__ __launch_bounds__(256) void cvt_x_kernel(const float* __restrict__ x,
                                                    __bf16* __restrict__ xb, int n4) {
    int i = blockIdx.x * blockDim.x + threadIdx.x;
    if (i >= n4) return;
    const float4 f = ((const float4*)x)[i];
    v4bf o;
    o[0] = (__bf16)f.x; o[1] = (__bf16)f.y; o[2] = (__bf16)f.z; o[3] = (__bf16)f.w;
    ((v4bf*)xb)[i] = o;
}

// ---------------------------------------------------------------------------
// Kernel 0b: W (C x H, f32) -> Wt (H x C, bf16)  (transpose + convert)
// ---------------------------------------------------------------------------
__global__ __launch_bounds__(256) void cvt_w_kernel(const float* __restrict__ w,
                                                    __bf16* __restrict__ wt, int n) {
    int o = blockIdx.x * blockDim.x + threadIdx.x;
    if (o >= n) return;
    int h = o / C_;
    int c = o - h * C_;
    wt[o] = (__bf16)w[c * H_ + h];
}

// ---------------------------------------------------------------------------
// Kernel 1: projections.  Q,K row-major bf16 [B*T][H]; V transposed [B][H][T].
// Grid = B*T/16 row tiles, 256 threads = 8 waves; wave w owns n-tile w.
// ---------------------------------------------------------------------------
__global__ __launch_bounds__(256) void proj_kernel(const __bf16* __restrict__ xb,
                                                   const __bf16* __restrict__ wt, // [3][H][C]
                                                   __bf16* __restrict__ Qb,
                                                   __bf16* __restrict__ Kb,
                                                   __bf16* __restrict__ Vtb) {
    const int lane = threadIdx.x & 31;
    const int wave = threadIdx.x >> 5;
    const int lo = lane & 15, hi = lane >> 4;
    const int n0 = wave * 16;
    const long t0 = (long)blockIdx.x * 16;

    const __bf16* xrow = xb + (t0 + lo) * C_;   // A-fragment row for this lane
    const int b = (int)(t0 / T_);
    const int tloc = (int)(t0 - (long)b * T_);

    for (int m = 0; m < 3; ++m) {
        const __bf16* wrow = wt + (size_t)m * H_ * C_ + (size_t)(n0 + lo) * C_;
        v8f acc = {0.f, 0.f, 0.f, 0.f, 0.f, 0.f, 0.f, 0.f};
#pragma unroll 4
        for (int kc = 0; kc < C_; kc += 32) {
            // A fragment (16x32 bf16)
            v8bf a0 = *(const v8bf*)(xrow + kc + hi * 16);
            v8bf a1 = *(const v8bf*)(xrow + kc + hi * 16 + 8);
            v16bf a = cat16(a0, a1);
            // B fragment (32x16 bf16): contiguous 32B of pre-transposed W
            v16bf bf = *(const v16bf*)(wrow + kc + hi * 16);
            acc = wmma_bf16(a, bf, acc);
        }
        const int col = n0 + lo;
        if (m == 0) {
#pragma unroll
            for (int r = 0; r < 8; ++r)
                Qb[(t0 + r + 8 * hi) * H_ + col] = (__bf16)acc[r];
        } else if (m == 1) {
#pragma unroll
            for (int r = 0; r < 8; ++r)
                Kb[(t0 + r + 8 * hi) * H_ + col] = (__bf16)acc[r];
        } else {
#pragma unroll
            for (int r = 0; r < 8; ++r)
                Vtb[((long)b * H_ + col) * T_ + tloc + r + 8 * hi] = (__bf16)acc[r];
        }
    }
}

// ---------------------------------------------------------------------------
// Kernel 2: fused causal flash attention.  One wave per 16-query tile.
// Processes keys in blocks of 32 (two 16x16 score WMMA tiles -> one 16x32 P
// A-fragment -> 8 PV WMMAs over H=128).
// ---------------------------------------------------------------------------
__global__ __launch_bounds__(32) void attn_kernel(const __bf16* __restrict__ Qb,
                                                  const __bf16* __restrict__ Kb,
                                                  const __bf16* __restrict__ Vtb,
                                                  float* __restrict__ out) {
    __shared__ __align__(32) __bf16 Pl[16 * 32];

    const int qt = blockIdx.x;
    const int b  = qt / (T_ / 16);
    const int q0 = (qt % (T_ / 16)) * 16;
    const int lane = threadIdx.x;
    const int lo = lane & 15, hi = lane >> 4;

    // Preload Q A-fragments for all 4 K-chunks of H=128
    const __bf16* qrow = Qb + ((long)b * T_ + q0 + lo) * H_;
    v16bf qa[4];
#pragma unroll
    for (int c = 0; c < 4; ++c) {
        const int kc = c * 32;
        v8bf a0 = *(const v8bf*)(qrow + kc + hi * 16);
        v8bf a1 = *(const v8bf*)(qrow + kc + hi * 16 + 8);
        qa[c] = cat16(a0, a1);
    }

    v8f o[8];
#pragma unroll
    for (int nt = 0; nt < 8; ++nt) o[nt] = (v8f){0.f,0.f,0.f,0.f,0.f,0.f,0.f,0.f};
    float mrow[8], lrow[8];
#pragma unroll
    for (int r = 0; r < 8; ++r) { mrow[r] = -3.0e38f; lrow[r] = 0.f; }

    const int send = q0 + 16;  // exclusive causal key bound for this tile
    for (int s0 = 0; s0 < send; s0 += 32) {
        // ---- scores: two 16x16 tiles over keys [s0, s0+32) ----
        v8f s0a = (v8f){0.f,0.f,0.f,0.f,0.f,0.f,0.f,0.f};
        v8f s1a = (v8f){0.f,0.f,0.f,0.f,0.f,0.f,0.f,0.f};
        const __bf16* k0row = Kb + ((long)b * T_ + s0 + lo) * H_;
        const __bf16* k1row = k0row + 16 * H_;
#pragma unroll
        for (int c = 0; c < 4; ++c) {
            const int kc = c * 32;
            v16bf kb0 = *(const v16bf*)(k0row + kc + hi * 16);
            s0a = wmma_bf16(qa[c], kb0, s0a);
            v16bf kb1 = *(const v16bf*)(k1row + kc + hi * 16);
            s1a = wmma_bf16(qa[c], kb1, s1a);
        }

        // ---- online softmax update (per-row, rows r+8*hi, col = lo) ----
#pragma unroll
        for (int r = 0; r < 8; ++r) {
            const int rowq = q0 + r + 8 * hi;
            float v0 = s0a[r] * SOFTMAX_SCALE;
            float v1 = s1a[r] * SOFTMAX_SCALE;
            v0 = (s0 + lo      > rowq) ? -3.0e38f : v0;   // causal mask
            v1 = (s0 + 16 + lo > rowq) ? -3.0e38f : v1;
            float mx = fmaxf(v0, v1);
#pragma unroll
            for (int d = 1; d < 16; d <<= 1) mx = fmaxf(mx, __shfl_xor(mx, d, 32));
            const float mnew  = fmaxf(mrow[r], mx);
            const float alpha = __expf(mrow[r] - mnew);
            const float p0 = __expf(v0 - mnew);
            const float p1 = __expf(v1 - mnew);
            float ps = p0 + p1;
#pragma unroll
            for (int d = 1; d < 16; d <<= 1) ps += __shfl_xor(ps, d, 32);
            lrow[r] = lrow[r] * alpha + ps;
            mrow[r] = mnew;
#pragma unroll
            for (int nt = 0; nt < 8; ++nt) o[nt][r] *= alpha;
            // stash P (C-layout) into LDS for relayout to A-fragment
            Pl[(r + 8 * hi) * 32 + lo]      = (__bf16)p0;
            Pl[(r + 8 * hi) * 32 + 16 + lo] = (__bf16)p1;
        }
        __syncthreads();  // single-wave WG: orders DS ops (compiler waits dscnt)

        // ---- P A-fragment (16 rows x 32 keys) from LDS ----
        const __bf16* prow = &Pl[lo * 32];
        v8bf pa0 = *(const v8bf*)(prow + hi * 16);
        v8bf pa1 = *(const v8bf*)(prow + hi * 16 + 8);
        v16bf pa = cat16(pa0, pa1);
        __syncthreads();

        // ---- O += P @ V over all 8 output n-tiles ----
#pragma unroll
        for (int nt = 0; nt < 8; ++nt) {
            const __bf16* vrow = Vtb + ((long)b * H_ + nt * 16 + lo) * T_ + s0;
            v16bf vb = *(const v16bf*)(vrow + hi * 16);
            o[nt] = wmma_bf16(pa, vb, o[nt]);
        }
    }

    // ---- normalize and store f32 output ----
    float inv[8];
#pragma unroll
    for (int r = 0; r < 8; ++r) inv[r] = 1.0f / lrow[r];
#pragma unroll
    for (int nt = 0; nt < 8; ++nt) {
#pragma unroll
        for (int r = 0; r < 8; ++r) {
            out[((long)b * T_ + q0 + r + 8 * hi) * H_ + nt * 16 + lo] = o[nt][r] * inv[r];
        }
    }
}

// ---------------------------------------------------------------------------
extern "C" void kernel_launch(void* const* d_in, const int* in_sizes, int n_in,
                              void* d_out, int out_size, void* d_ws, size_t ws_size,
                              hipStream_t stream) {
    const float* x  = (const float*)d_in[0];
    const float* Wq = (const float*)d_in[1];
    const float* Wk = (const float*)d_in[2];
    const float* Wv = (const float*)d_in[3];

    char* ws = (char*)d_ws;
    size_t off = 0;
    __bf16* xb  = (__bf16*)(ws + off); off += (size_t)B_ * T_ * C_ * sizeof(__bf16);
    __bf16* Wt  = (__bf16*)(ws + off); off += (size_t)3 * H_ * C_ * sizeof(__bf16);
    __bf16* Qb  = (__bf16*)(ws + off); off += (size_t)B_ * T_ * H_ * sizeof(__bf16);
    __bf16* Kb  = (__bf16*)(ws + off); off += (size_t)B_ * T_ * H_ * sizeof(__bf16);
    __bf16* Vtb = (__bf16*)(ws + off); off += (size_t)B_ * T_ * H_ * sizeof(__bf16);

    {   // x -> bf16
        const int n4 = B_ * T_ * C_ / 4;
        cvt_x_kernel<<<(n4 + 255) / 256, 256, 0, stream>>>(x, xb, n4);
    }
    {   // W -> transposed bf16
        const int n = C_ * H_;
        cvt_w_kernel<<<(n + 255) / 256, 256, 0, stream>>>(Wq, Wt + 0 * (size_t)H_ * C_, n);
        cvt_w_kernel<<<(n + 255) / 256, 256, 0, stream>>>(Wk, Wt + 1 * (size_t)H_ * C_, n);
        cvt_w_kernel<<<(n + 255) / 256, 256, 0, stream>>>(Wv, Wt + 2 * (size_t)H_ * C_, n);
    }
    // Q/K/V projections via WMMA
    proj_kernel<<<B_ * T_ / 16, 256, 0, stream>>>(xb, Wt, Qb, Kb, Vtb);
    // fused causal flash attention via WMMA
    attn_kernel<<<B_ * T_ / 16, 32, 0, stream>>>(Qb, Kb, Vtb, (float*)d_out);
}
